// AdjunctiveClassifier_85736137163523
// MI455X (gfx1250) — matrix operand order repaired and checked by reading, not compiled, not run on hardware
//
#include <hip/hip_runtime.h>

#define N_PATCHES 131072
#define N_BAGS    128
#define IN_DIM    1024
#define HID_DIM   512
#define PLANE_DW  262144   // (1024/32 kt) * (512/16 nt) * 32 lanes * 8 dwords

typedef __attribute__((ext_vector_type(16))) __bf16   v16bf;
typedef __attribute__((ext_vector_type(8)))  float    v8f;
typedef __attribute__((ext_vector_type(4)))  float    f4;
typedef __attribute__((ext_vector_type(4)))  unsigned u4;

union BF16Frag { v16bf v; unsigned d[8]; u4 q[2]; };

__device__ __forceinline__ float bf16_hi_part(float f) {
    union { float f; unsigned u; } c; c.f = f;
    c.u &= 0xFFFF0000u;
    return c.f;
}
// Pack truncated-bf16 of (a,b) into one dword: a -> bits[15:0], b -> bits[31:16].
// Single v_perm_b32 selecting the high halves of both floats.
__device__ __forceinline__ unsigned pack_bf16_hi(float a, float b) {
    return __builtin_amdgcn_perm(__float_as_uint(b), __float_as_uint(a), 0x07060302u);
}

// ---------------------------------------------------------------------------
// Prep: swizzle W1 (1024x512 f32, row-major K x N) into WMMA B-fragment
// layout, split into bf16 hi/lo planes. Layout per (kt, nt) 32x16 tile:
//   lanes 0-15: N = lane,    K = kt*32 + 2v + {0,1}
//   lanes16-31: N = lane-16, K = kt*32 + 16 + 2v + {0,1}
// dword index = ((kt*32 + nt)*32 + lane)*8 + v ; plane1 (lo) at +PLANE_DW.
// Also zero the bag-logits region of d_out (atomically accumulated later).
// ---------------------------------------------------------------------------
__global__ __launch_bounds__(256) void prep_kernel(const float* __restrict__ W1,
                                                   unsigned* __restrict__ bws,
                                                   float* __restrict__ out) {
    int tid = blockIdx.x * 256 + threadIdx.x;     // 0 .. 262143
    if (tid < 2 * N_BAGS) out[tid] = 0.0f;

    int v    = tid & 7;
    int lane = (tid >> 3) & 31;
    int nt   = (tid >> 8) & 31;
    int kt   = (tid >> 13) & 31;

    int n = nt * 16 + (lane & 15);
    int k = kt * 32 + ((lane >> 4) << 4) + 2 * v;

    float w0 = W1[(size_t)k * HID_DIM + n];
    float w1 = W1[(size_t)(k + 1) * HID_DIM + n];
    float h0 = bf16_hi_part(w0), h1 = bf16_hi_part(w1);

    size_t idx = (((size_t)kt * 32 + nt) * 32 + lane) * 8 + v;
    bws[idx]            = pack_bf16_hi(w0, w1);
    bws[PLANE_DW + idx] = pack_bf16_hi(w0 - h0, w1 - h1);
}

// ---------------------------------------------------------------------------
// Fused: h = relu(feat @ W1 + b1); logits = h @ W2 + b2; evidence; bag sums.
// Block: 256 threads = 8 waves; tile = 64 rows x 512 cols.
// Wave grid: 2 (M) x 4 (N); wave tile = 32 rows x 128 cols = 2x8 WMMA tiles.
// bf16x3 split-precision accumulate in f32 (error ~2^-16 relative).
// A tile is software-pipelined: kt+1 raw fp32 loads issue before kt's WMMAs.
// ---------------------------------------------------------------------------
__global__ __launch_bounds__(256) void mil_fused_kernel(
    const float* __restrict__ feat, const float* __restrict__ att,
    const unsigned* __restrict__ bws,
    const float* __restrict__ b1, const float* __restrict__ W2,
    const float* __restrict__ b2, float* __restrict__ out) {

    const int tid   = threadIdx.x;
    const int lane  = tid & 31;
    const int wave  = tid >> 5;
    const int mwave = wave >> 2;          // 0..1
    const int nwave = wave & 3;           // 0..3
    const int half  = lane >> 4;          // 0/1
    const int l16   = lane & 15;
    const int m0    = blockIdx.x * 64;
    const int rowb  = m0 + mwave * 32 + l16;

    const float* apbase[2];
    apbase[0] = feat + (size_t)(rowb) * IN_DIM + half * 16;
    apbase[1] = feat + (size_t)(rowb + 16) * IN_DIM + half * 16;

    const v8f zero = {0.f,0.f,0.f,0.f,0.f,0.f,0.f,0.f};
    v8f acc[2][8];
    #pragma unroll
    for (int mt = 0; mt < 2; ++mt)
        #pragma unroll
        for (int nt = 0; nt < 8; ++nt) acc[mt][nt] = zero;

    // ---- software-pipelined raw A tile (fp32, 16 values/lane/mt)
    f4 raw[2][4], rawn[2][4];
    #pragma unroll
    for (int mt = 0; mt < 2; ++mt) {
        const float* ap = apbase[mt];
        raw[mt][0] = *(const f4*)(ap);
        raw[mt][1] = *(const f4*)(ap + 4);
        raw[mt][2] = *(const f4*)(ap + 8);
        raw[mt][3] = *(const f4*)(ap + 12);
    }

    for (int kt = 0; kt < 32; ++kt) {
        // ---- prefetch next kt's raw A tile (completes under the WMMAs below)
        if (kt < 31) {
            const int k0n = (kt + 1) * 32;
            #pragma unroll
            for (int mt = 0; mt < 2; ++mt) {
                const float* ap = apbase[mt] + k0n;
                rawn[mt][0] = *(const f4*)(ap);
                rawn[mt][1] = *(const f4*)(ap + 4);
                rawn[mt][2] = *(const f4*)(ap + 8);
                rawn[mt][3] = *(const f4*)(ap + 12);
            }
        }

        // ---- split current raw A into bf16 hi/lo fragments (perm-packed)
        BF16Frag Ahi[2], Alo[2];
        #pragma unroll
        for (int mt = 0; mt < 2; ++mt) {
            float fl[16] = { raw[mt][0].x, raw[mt][0].y, raw[mt][0].z, raw[mt][0].w,
                             raw[mt][1].x, raw[mt][1].y, raw[mt][1].z, raw[mt][1].w,
                             raw[mt][2].x, raw[mt][2].y, raw[mt][2].z, raw[mt][2].w,
                             raw[mt][3].x, raw[mt][3].y, raw[mt][3].z, raw[mt][3].w };
            #pragma unroll
            for (int i = 0; i < 8; ++i) {
                float a  = fl[2 * i], b = fl[2 * i + 1];
                float ah = bf16_hi_part(a), bh = bf16_hi_part(b);
                Ahi[mt].d[i] = pack_bf16_hi(a, b);
                Alo[mt].d[i] = pack_bf16_hi(a - ah, b - bh);
            }
        }

        // ---- B fragments from pre-swizzled workspace; 3-product accumulate
        #pragma unroll
        for (int nt = 0; nt < 8; ++nt) {
            size_t bidx = (((size_t)kt * 32 + (nwave * 8 + nt)) * 32 + lane) * 8;
            BF16Frag Bhi, Blo;
            Bhi.q[0] = *(const u4*)(bws + bidx);
            Bhi.q[1] = *(const u4*)(bws + bidx + 4);
            Blo.q[0] = *(const u4*)(bws + PLANE_DW + bidx);
            Blo.q[1] = *(const u4*)(bws + PLANE_DW + bidx + 4);
            #pragma unroll
            for (int mt = 0; mt < 2; ++mt) {
                acc[mt][nt] = __builtin_amdgcn_wmma_f32_16x16x32_bf16(
                    false, Ahi[mt].v, false, Bhi.v, (short)0, acc[mt][nt], false, false);
                acc[mt][nt] = __builtin_amdgcn_wmma_f32_16x16x32_bf16(
                    false, Ahi[mt].v, false, Blo.v, (short)0, acc[mt][nt], false, false);
                acc[mt][nt] = __builtin_amdgcn_wmma_f32_16x16x32_bf16(
                    false, Alo[mt].v, false, Bhi.v, (short)0, acc[mt][nt], false, false);
            }
        }

        // ---- rotate pipeline registers
        #pragma unroll
        for (int mt = 0; mt < 2; ++mt)
            #pragma unroll
            for (int i = 0; i < 4; ++i) raw[mt][i] = rawn[mt][i];
    }

    // ---- epilogue: +b1, relu, project to 2 logits, reduce over hidden dim
    __shared__ float logits[64][2];
    __shared__ float bagsum[2];
    if (tid < 128) ((float*)logits)[tid] = 0.0f;
    if (tid < 2)   bagsum[tid] = 0.0f;
    __syncthreads();

    float po[2][8][2];
    #pragma unroll
    for (int mt = 0; mt < 2; ++mt)
        #pragma unroll
        for (int r = 0; r < 8; ++r) { po[mt][r][0] = 0.f; po[mt][r][1] = 0.f; }

    #pragma unroll
    for (int nt = 0; nt < 8; ++nt) {
        int n = nwave * 128 + nt * 16 + l16;
        float b1v = b1[n];
        float w20 = W2[2 * n], w21 = W2[2 * n + 1];
        #pragma unroll
        for (int mt = 0; mt < 2; ++mt)
            #pragma unroll
            for (int r = 0; r < 8; ++r) {
                float h = fmaxf(acc[mt][nt][r] + b1v, 0.0f);
                po[mt][r][0] += h * w20;
                po[mt][r][1] += h * w21;
            }
    }

    // butterfly-reduce across the 16 N-lanes of each half-wave
    #pragma unroll
    for (int mask = 1; mask < 16; mask <<= 1)
        #pragma unroll
        for (int mt = 0; mt < 2; ++mt)
            #pragma unroll
            for (int r = 0; r < 8; ++r) {
                po[mt][r][0] += __shfl_xor(po[mt][r][0], mask, 32);
                po[mt][r][1] += __shfl_xor(po[mt][r][1], mask, 32);
            }

    if (l16 == 0) {
        #pragma unroll
        for (int mt = 0; mt < 2; ++mt)
            #pragma unroll
            for (int r = 0; r < 8; ++r) {
                int row = mwave * 32 + mt * 16 + half * 8 + r;
                atomicAdd(&logits[row][0], po[mt][r][0]);
                atomicAdd(&logits[row][1], po[mt][r][1]);
            }
    }
    __syncthreads();

    // ---- evidence + per-bag accumulation (block's 64 rows lie in one 1024-row bag)
    if (tid < 128) {
        int row = tid >> 1, o = tid & 1;
        size_t p = (size_t)m0 + row;
        float l  = logits[row][o] + b2[o];
        float pe = att[2 * p + 0] * fmaxf(l, 0.0f);
        float ne = att[2 * p + 1] * fmaxf(-l, 0.0f);
        float net = pe - ne;
        out[256 + 2 * p + o]                          = pe;
        out[256 + 2 * (size_t)N_PATCHES + 2 * p + o]  = ne;
        out[256 + 4 * (size_t)N_PATCHES + 2 * p + o]  = net;
        atomicAdd(&bagsum[o], net);
    }
    __syncthreads();
    if (tid < 2) {
        int bag = m0 >> 10;   // N_PATCHES / N_BAGS = 1024 patches per bag
        atomicAdd(&out[2 * bag + tid], bagsum[tid]);
    }
}

extern "C" void kernel_launch(void* const* d_in, const int* in_sizes, int n_in,
                              void* d_out, int out_size, void* d_ws, size_t ws_size,
                              hipStream_t stream) {
    (void)in_sizes; (void)n_in; (void)out_size; (void)ws_size;
    const float* feat = (const float*)d_in[0];
    const float* att  = (const float*)d_in[1];
    // d_in[2] bag_sizes: uniform N_PATCHES/N_BAGS by construction
    const float* W1   = (const float*)d_in[3];
    const float* b1   = (const float*)d_in[4];
    const float* W2   = (const float*)d_in[5];
    const float* b2   = (const float*)d_in[6];
    float*    out = (float*)d_out;
    unsigned* bws = (unsigned*)d_ws;   // needs 2 MB scratch

    prep_kernel<<<1024, 256, 0, stream>>>(W1, bws, out);
    mil_fused_kernel<<<N_PATCHES / 64, 256, 0, stream>>>(feat, att, bws, b1, W2, b2, out);
}
